// ProNet_4922032521430
// MI455X (gfx1250) — matrix operand order repaired and checked by reading, not compiled, not run on hardware
//
#include <hip/hip_runtime.h>
#include <stddef.h>

// ---------------------------------------------------------------------------
// ProNet forward for MI455X (gfx1250, wave32).
// Dense layers via v_wmma_f32_16x16x32_bf16 (bf16 in, f32 accum).
// Each wave computes a 16x64 output strip: 1 A fragment feeds 4 independent
// WMMAs per k-step (amortizes A loads 4x and hides the WMMA->WMMA RAW hazard
// without v_nops). All GEMM K dims are multiples of 32 by construction
// (K=12/16 inputs/weights zero-padded to 32 once), so fragment loads are
// unguarded vectorized b128 loads. Edge conv via atomic scatter (mem bound).
// ---------------------------------------------------------------------------

typedef __attribute__((ext_vector_type(16))) __bf16 v16bf;
typedef __attribute__((ext_vector_type(8)))  __bf16 bf16x8;
typedef __attribute__((ext_vector_type(8)))  float  v8f;
typedef __attribute__((ext_vector_type(4)))  float  f32x4;

#define ACT_NONE 0
#define ACT_SILU 1
#define ACT_RELU 2

// ---- WMMA fragment loaders (layouts per CDNA5 ISA §7.12.2) -----
// A 16x32 bf16: lanes 0-15 -> M=lane, K strips {0..7}(v0-3) and {16..23}(v4-7)
//               lanes 16-31 -> same M, K strips {8..15} and {24..31}
// k0 = kBase + (lane>>4)*8; r[0..7] = p[k0..k0+7]; r[8..15] = p[k0+16..k0+23]
__device__ __forceinline__ v16bf fragA(const float* __restrict__ p, int k0) {
  f32x4 a0 = *(const f32x4*)(p + k0);
  f32x4 a1 = *(const f32x4*)(p + k0 + 4);
  f32x4 a2 = *(const f32x4*)(p + k0 + 16);
  f32x4 a3 = *(const f32x4*)(p + k0 + 20);
  v16bf r;
#pragma unroll
  for (int i = 0; i < 4; ++i) {
    r[i]      = (__bf16)a0[i];
    r[i + 4]  = (__bf16)a1[i];
    r[i + 8]  = (__bf16)a2[i];
    r[i + 12] = (__bf16)a3[i];
  }
  return r;
}

__device__ __forceinline__ v16bf fragA(const __bf16* __restrict__ p, int k0) {
  bf16x8 a0 = *(const bf16x8*)(p + k0);
  bf16x8 a1 = *(const bf16x8*)(p + k0 + 16);
  v16bf r;
#pragma unroll
  for (int i = 0; i < 8; ++i) { r[i] = a0[i]; r[i + 8] = a1[i]; }
  return r;
}

// B 32x16 bf16 (K x N): lanes 0-15 -> N=lane, K strips {0..7} and {16..23}
//                       lanes 16-31 -> N=lane-16, K strips {8..15} and {24..31}
// Since y = x @ W^T, B[k][n] = W[n][k]: two 8-k runs (k0..k0+7, k0+16..k0+23) from row n of W.
__device__ __forceinline__ v16bf fragB(const __bf16* __restrict__ p, int k0) {
  bf16x8 b0 = *(const bf16x8*)(p + k0);
  bf16x8 b1 = *(const bf16x8*)(p + k0 + 16);
  v16bf r;
#pragma unroll
  for (int i = 0; i < 8; ++i) { r[i] = b0[i]; r[i + 8] = b1[i]; }
  return r;
}

__device__ __forceinline__ void epilogue_tile(v8f acc, int row0, int col,
                                              const float* __restrict__ bias,
                                              const float* __restrict__ resid,
                                              int ldr,
                                              float* __restrict__ Cf,
                                              __bf16* __restrict__ Cb,
                                              int ldc, int act) {
  float bv = bias ? bias[col] : 0.0f;
#pragma unroll
  for (int v = 0; v < 8; ++v) {
    float x = acc[v] + bv;
    if (act == ACT_SILU)      x = x * (1.0f / (1.0f + __expf(-x)));
    else if (act == ACT_RELU) x = fmaxf(x, 0.0f);
    size_t row = (size_t)(row0 + v);
    if (resid) x += resid[row * ldr + col];
    if (Cf) Cf[row * ldc + col] = x;
    else    Cb[row * ldc + col] = (__bf16)x;
  }
}

// ---- Generic fused GEMM: C = act(A1*B1^T [+ A2*B2^T] + bias) [+ resid] ----
// One wave computes one 16x64 output strip (4 accumulators) over full K.
// 8 waves / block. Requires: M % 16 == 0, Nout % 64 == 0, K % 32 == 0.
template <typename AT>
__global__ void pronet_gemm_wmma(const AT* __restrict__ A1, int lda1, int K1,
                                 const AT* __restrict__ A2, int lda2, int K2,
                                 const __bf16* __restrict__ B1, int ldb1,
                                 const __bf16* __restrict__ B2, int ldb2,
                                 const float* __restrict__ bias,
                                 const float* __restrict__ resid, int ldr,
                                 float* __restrict__ Cf,
                                 __bf16* __restrict__ Cb, int ldc,
                                 int M, int Nout, int act) {
  int wave = threadIdx.x >> 5;
  int lane = threadIdx.x & 31;
  int hlf  = lane >> 4;
  int l15  = lane & 15;
  int nStr = Nout >> 6;                          // 64-wide strips
  int strip = blockIdx.x * (blockDim.x >> 5) + wave;
  if (strip >= (M >> 4) * nStr) return;          // wave-uniform: EXEC all-1s
  int mT = strip / nStr;
  int nS = strip - mT * nStr;

  int kA = hlf * 8;    // A lane-half K offset
  int kB = hlf * 8;    // B lane-half K offset
  const AT*     aRow = A1 + (size_t)(mT * 16 + l15) * lda1;
  const __bf16* bRow = B1 + (size_t)(nS * 64 + l15) * ldb1;
  v8f acc0 = {}, acc1 = {}, acc2 = {}, acc3 = {};
  for (int kb = 0; kb < K1; kb += 32) {
    v16bf a  = fragA(aRow, kb + kA);
    v16bf b0 = fragB(bRow,                       kb + kB);
    v16bf b1 = fragB(bRow + (size_t)16 * ldb1,   kb + kB);
    v16bf b2 = fragB(bRow + (size_t)32 * ldb1,   kb + kB);
    v16bf b3 = fragB(bRow + (size_t)48 * ldb1,   kb + kB);
    acc0 = __builtin_amdgcn_wmma_f32_16x16x32_bf16(false, a, false, b0,
                                                   (short)0, acc0, false, false);
    acc1 = __builtin_amdgcn_wmma_f32_16x16x32_bf16(false, a, false, b1,
                                                   (short)0, acc1, false, false);
    acc2 = __builtin_amdgcn_wmma_f32_16x16x32_bf16(false, a, false, b2,
                                                   (short)0, acc2, false, false);
    acc3 = __builtin_amdgcn_wmma_f32_16x16x32_bf16(false, a, false, b3,
                                                   (short)0, acc3, false, false);
  }
  if (A2) {
    const AT*     aRow2 = A2 + (size_t)(mT * 16 + l15) * lda2;
    const __bf16* bRow2 = B2 + (size_t)(nS * 64 + l15) * ldb2;
    for (int kb = 0; kb < K2; kb += 32) {
      v16bf a  = fragA(aRow2, kb + kA);
      v16bf b0 = fragB(bRow2,                     kb + kB);
      v16bf b1 = fragB(bRow2 + (size_t)16 * ldb2, kb + kB);
      v16bf b2 = fragB(bRow2 + (size_t)32 * ldb2, kb + kB);
      v16bf b3 = fragB(bRow2 + (size_t)48 * ldb2, kb + kB);
      acc0 = __builtin_amdgcn_wmma_f32_16x16x32_bf16(false, a, false, b0,
                                                     (short)0, acc0, false, false);
      acc1 = __builtin_amdgcn_wmma_f32_16x16x32_bf16(false, a, false, b1,
                                                     (short)0, acc1, false, false);
      acc2 = __builtin_amdgcn_wmma_f32_16x16x32_bf16(false, a, false, b2,
                                                     (short)0, acc2, false, false);
      acc3 = __builtin_amdgcn_wmma_f32_16x16x32_bf16(false, a, false, b3,
                                                     (short)0, acc3, false, false);
    }
  }

  // C layout: lane (0-15): N=lane, M=vgpr; lane (16-31): N=lane-16, M=vgpr+8
  int row0 = mT * 16 + hlf * 8;
  int col0 = nS * 64 + l15;
  epilogue_tile(acc0, row0, col0,      bias, resid, ldr, Cf, Cb, ldc, act);
  epilogue_tile(acc1, row0, col0 + 16, bias, resid, ldr, Cf, Cb, ldc, act);
  epilogue_tile(acc2, row0, col0 + 32, bias, resid, ldr, Cf, Cb, ldc, act);
  epilogue_tile(acc3, row0, col0 + 48, bias, resid, ldr, Cf, Cb, ldc, act);
}

// ---- fp32 -> bf16 conversion (flat) ---------------------------------------
__global__ void pronet_cvt_bf16(const float* __restrict__ src,
                                __bf16* __restrict__ dst, long n) {
  long i = (long)blockIdx.x * blockDim.x + threadIdx.x;
  for (; i < n; i += (long)gridDim.x * blockDim.x) dst[i] = (__bf16)src[i];
}

// ---- fp32 [M,Kin] -> bf16 [M,Kpad] with zero K-padding ---------------------
__global__ void pronet_pad_cvt(const float* __restrict__ src,
                               __bf16* __restrict__ dst,
                               int Kin, int Kpad, long M) {
  long total = M * (long)Kpad;
  long i = (long)blockIdx.x * blockDim.x + threadIdx.x;
  for (; i < total; i += (long)gridDim.x * blockDim.x) {
    long n = i / Kpad;
    int  c = (int)(i - n * Kpad);
    dst[i] = (c < Kin) ? (__bf16)src[n * Kin + c] : (__bf16)0.0f;
  }
}

__global__ void pronet_fill(float* __restrict__ p, float v, long n) {
  long i = (long)blockIdx.x * blockDim.x + threadIdx.x;
  for (; i < n; i += (long)gridDim.x * blockDim.x) p[i] = v;
}

// ---- build x_in = [ndata | aa_emb[x_aa]] and stash aa ----------------------
__global__ void pronet_embed(const float* __restrict__ ndata,
                             const float* __restrict__ aa_emb,
                             const int* __restrict__ x_aa,
                             float* __restrict__ xin,
                             float* __restrict__ aaBuf, int Nn) {
  int i = blockIdx.x * blockDim.x + threadIdx.x;
  if (i >= Nn * 128) return;
  int n = i >> 7, c = i & 127;
  float v;
  if (c < 64) {
    v = ndata[(size_t)n * 64 + c];
  } else {
    v = aa_emb[(size_t)x_aa[n] * 64 + (c - 64)];
    aaBuf[(size_t)n * 64 + (c - 64)] = v;
  }
  xin[i] = v;
}

// ---- LayerNorm over 128 channels: one wave32 per node ----------------------
__global__ void pronet_layernorm(float* __restrict__ x,
                                 const float* __restrict__ g,
                                 const float* __restrict__ b, int Nn) {
  int wave = threadIdx.x >> 5, lane = threadIdx.x & 31;
  int n = blockIdx.x * (blockDim.x >> 5) + wave;
  if (n >= Nn) return;
  float* p = x + (size_t)n * 128;
  float vals[4], s = 0.f, s2 = 0.f;
#pragma unroll
  for (int j = 0; j < 4; ++j) {
    float v = p[lane + 32 * j];
    vals[j] = v; s += v; s2 += v * v;
  }
#pragma unroll
  for (int off = 16; off > 0; off >>= 1) {
    s  += __shfl_xor(s,  off, 32);
    s2 += __shfl_xor(s2, off, 32);
  }
  float mu  = s * (1.0f / 128.0f);
  float var = s2 * (1.0f / 128.0f) - mu * mu;
  float inv = rsqrtf(var + 1e-5f);
#pragma unroll
  for (int j = 0; j < 4; ++j) {
    int c = lane + 32 * j;
    p[c] = (vals[j] - mu) * inv * g[c] + b[c];
  }
}

// ---- EdgeGraphConv scatter: agg[dst] += x1[src] * f (atomic, mem bound) ----
__global__ void pronet_conv_scatter(const float* __restrict__ x1,
                                    const __bf16* __restrict__ f,
                                    const int* __restrict__ src,
                                    const int* __restrict__ dst,
                                    float* __restrict__ agg, long total) {
  long i = (long)blockIdx.x * blockDim.x + threadIdx.x;
  for (; i < total; i += (long)gridDim.x * blockDim.x) {
    long e = i >> 7;
    int  c = (int)(i & 127);
    float v = x1[(size_t)src[e] * 128 + c] * (float)f[i];
    atomicAdd(&agg[(size_t)dst[e] * 128 + c], v);
  }
}

// ---- final out = h @ W_out^T + b (Nout = 1): one wave32 per node -----------
__global__ void pronet_outdot(const float* __restrict__ h,
                              const float* __restrict__ W,
                              const float* __restrict__ b,
                              float* __restrict__ out, int Nn) {
  int wave = threadIdx.x >> 5, lane = threadIdx.x & 31;
  int n = blockIdx.x * (blockDim.x >> 5) + wave;
  if (n >= Nn) return;
  const float* p = h + (size_t)n * 128;
  float s = 0.f;
#pragma unroll
  for (int j = 0; j < 4; ++j) { int c = lane + 32 * j; s += p[c] * W[c]; }
#pragma unroll
  for (int off = 16; off > 0; off >>= 1) s += __shfl_xor(s, off, 32);
  if (lane == 0) out[n] = s + b[0];
}

// ---------------------------------------------------------------------------
// Host orchestration
// ---------------------------------------------------------------------------
static inline size_t alignUp(size_t x, size_t a) { return (x + a - 1) / a * a; }

template <typename AT>
static inline void launch_gemm(hipStream_t s,
                               const AT* A1, int lda1, int K1,
                               const AT* A2, int lda2, int K2,
                               const void* B1, int ldb1,
                               const void* B2, int ldb2,
                               const float* bias, const float* resid, int ldr,
                               float* Cf, void* Cb, int ldc,
                               int M, int Nout, int act) {
  int strips = (M / 16) * (Nout / 64);
  int blocks = (strips + 7) / 8;
  pronet_gemm_wmma<AT><<<blocks, 256, 0, s>>>(A1, lda1, K1, A2, lda2, K2,
                                              (const __bf16*)B1, ldb1,
                                              (const __bf16*)B2, ldb2,
                                              bias, resid, ldr,
                                              Cf, (__bf16*)Cb, ldc, M, Nout, act);
}

extern "C" void kernel_launch(void* const* d_in, const int* in_sizes, int n_in,
                              void* d_out, int out_size, void* d_ws, size_t ws_size,
                              hipStream_t stream) {
  const int Nn = in_sizes[0];          // 50000 (multiple of 16)
  const int Ee = in_sizes[2] / 12;     // 400000 (multiple of 16)

  const int*   x_aa   = (const int*)d_in[0];
  const float* ndata  = (const float*)d_in[1];
  const float* feat1  = (const float*)d_in[2];
  const float* posemb = (const float*)d_in[3];
  const int*   eidx   = (const int*)d_in[4];
  const int*   esrc   = eidx;
  const int*   edst   = eidx + Ee;

  // Param leaf indices: setup_inputs() dict insertion order.
  const int AA_EMB = 5, NDW = 6, NDB = 7, LNG = 8, LNB = 9;
  const int BLK0 = 10;                 // 30 leaves per block, 4 blocks
  const int VARW = 130, VARB = 131, OUTW = 132, OUTB = 133;
  (void)n_in;

  // ---- workspace bump allocator ----
  char* ws = (char*)d_ws;
  size_t off = 0;
  auto alloc = [&](size_t bytes) -> void* {
    void* p = ws + off;
    off = alignUp(off + bytes, 256);
    return p;
  };

  // ---- convert all GEMM weights to bf16 once (tiny: ~2.3 MB total) ----
  auto convW = [&](int idx, long nelem) -> void* {
    void* dst = alloc((size_t)nelem * 2);
    pronet_cvt_bf16<<<(int)((nelem + 255) / 256), 256, 0, stream>>>(
        (const float*)d_in[idx], (__bf16*)dst, nelem);
    return dst;
  };
  // K-padded weight conversion: [rows,Kin] -> [rows,32]
  auto convWpad = [&](int idx, int rows, int Kin) -> void* {
    long nelem = (long)rows * 32;
    void* dst = alloc((size_t)nelem * 2);
    pronet_pad_cvt<<<(int)((nelem + 255) / 256), 256, 0, stream>>>(
        (const float*)d_in[idx], (__bf16*)dst, Kin, 32, rows);
    return dst;
  };

  void* ndWb  = convW(NDW, 128 * 128);
  void* varWb = convW(VARW, 128 * 192);

  struct BW {
    void *lin1W, *lin2W, *f1W1, *f1W2, *f2W1, *f2W2;
    void *c1Wl, *c1Wr, *c2Wl, *c2Wr, *l1W, *l2W;
    void *cat0W, *cat1W, *cat2W, *ls0W, *ls1W, *finW;
    const float *lin1b, *lin2b, *c1bl, *c2bl, *l1b, *l2b;
    const float *cat0b, *cat1b, *cat2b, *ls0b, *ls1b, *finb;
  } bw[4];

  for (int k = 0; k < 4; ++k) {
    int p = BLK0 + k * 30;
    bw[k].lin1W = convW(p + 0, 128 * 128);  bw[k].lin1b = (const float*)d_in[p + 1];
    bw[k].lin2W = convW(p + 2, 128 * 128);  bw[k].lin2b = (const float*)d_in[p + 3];
    bw[k].f1W1  = convWpad(p + 4, 64, 12);            // [64,12] -> [64,32]
    bw[k].f1W2  = convW(p + 5, 128 * 64);
    bw[k].f2W1  = convWpad(p + 6, 64, 16);            // [64,16] -> [64,32]
    bw[k].f2W2  = convW(p + 7, 128 * 64);
    bw[k].c1Wl  = convW(p + 8, 128 * 128);  bw[k].c1bl  = (const float*)d_in[p + 9];
    bw[k].c1Wr  = convW(p + 10, 128 * 128);
    bw[k].c2Wl  = convW(p + 11, 128 * 128); bw[k].c2bl  = (const float*)d_in[p + 12];
    bw[k].c2Wr  = convW(p + 13, 128 * 128);
    bw[k].l1W   = convW(p + 14, 128 * 128); bw[k].l1b   = (const float*)d_in[p + 15];
    bw[k].l2W   = convW(p + 16, 128 * 128); bw[k].l2b   = (const float*)d_in[p + 17];
    bw[k].cat0W = convW(p + 18, 128 * 256); bw[k].cat0b = (const float*)d_in[p + 19];
    bw[k].cat1W = convW(p + 20, 128 * 128); bw[k].cat1b = (const float*)d_in[p + 21];
    bw[k].cat2W = convW(p + 22, 128 * 128); bw[k].cat2b = (const float*)d_in[p + 23];
    bw[k].ls0W  = convW(p + 24, 128 * 128); bw[k].ls0b  = (const float*)d_in[p + 25];
    bw[k].ls1W  = convW(p + 26, 128 * 128); bw[k].ls1b  = (const float*)d_in[p + 27];
    bw[k].finW  = convW(p + 28, 128 * 128); bw[k].finb  = (const float*)d_in[p + 29];
  }

  // ---- activation buffers ----
  size_t nodeB = (size_t)Nn * 128 * sizeof(float);
  float*  xA    = (float*)alloc(nodeB);
  float*  xB    = (float*)alloc(nodeB);
  float*  x1b   = (float*)alloc(nodeB);
  float*  x2b   = (float*)alloc(nodeB);
  float*  aggb  = (float*)alloc(nodeB);
  float*  yb    = (float*)alloc(nodeB);
  float*  h1b   = (float*)alloc(nodeB);
  float*  h2b   = (float*)alloc(nodeB);
  float*  aaBuf = (float*)alloc((size_t)Nn * 64 * sizeof(float));
  __bf16* f1pad = (__bf16*)alloc((size_t)Ee * 32 * 2);   // feature1 zero-padded
  __bf16* f2pad = (__bf16*)alloc((size_t)Ee * 32 * 2);   // pos_emb  zero-padded
  __bf16* midE  = (__bf16*)alloc((size_t)Ee * 64 * 2);   // edge two-linear mid
  __bf16* fE    = (__bf16*)alloc((size_t)Ee * 128 * 2);  // edge features
  if (off > ws_size) return;  // workspace too small; bail without OOB writes

  const float* lnG = (const float*)d_in[LNG];
  const float* lnB = (const float*)d_in[LNB];

  // ---- pad loop-invariant edge inputs once: [E,12/16] f32 -> [E,32] bf16 ----
  {
    long tot = (long)Ee * 32;
    int  blk = (int)((tot + 255) / 256);
    pronet_pad_cvt<<<blk, 256, 0, stream>>>(feat1,  f1pad, 12, 32, Ee);
    pronet_pad_cvt<<<blk, 256, 0, stream>>>(posemb, f2pad, 16, 32, Ee);
  }

  // ---- prologue: x = LayerNorm(Linear([ndata|aa])) ----
  pronet_embed<<<(Nn * 128 + 255) / 256, 256, 0, stream>>>(
      ndata, (const float*)d_in[AA_EMB], x_aa, yb, aaBuf, Nn);
  launch_gemm<float>(stream, yb, 128, 128, nullptr, 0, 0, ndWb, 128, nullptr, 0,
                     (const float*)d_in[NDB], nullptr, 0, xA, nullptr, 128,
                     Nn, 128, ACT_NONE);
  pronet_layernorm<<<(Nn + 7) / 8, 256, 0, stream>>>(xA, lnG, lnB, Nn);

  float* xcur = xA;
  float* xnxt = xB;
  const long convTot = (long)Ee * 128;
  const int  convBlk = (int)((convTot + 255) / 256);
  const long aggN    = (long)Nn * 128;
  const int  fillBlk = (int)((aggN + 255) / 256);

  // ---- interaction blocks ----
  for (int k = 0; k < 4; ++k) {
    const BW& w = bw[k];
    // x1 = silu(lin_1(x)); x2 = silu(lin_2(x))
    launch_gemm<float>(stream, xcur, 128, 128, nullptr, 0, 0, w.lin1W, 128,
                       nullptr, 0, w.lin1b, nullptr, 0, x1b, nullptr, 128,
                       Nn, 128, ACT_SILU);
    launch_gemm<float>(stream, xcur, 128, 128, nullptr, 0, 0, w.lin2W, 128,
                       nullptr, 0, w.lin2b, nullptr, 0, x2b, nullptr, 128,
                       Nn, 128, ACT_SILU);

    // f1 = feature1 @ W1^T @ W2^T  (E x 32pad -> 64 -> 128, bf16 pipeline)
    launch_gemm<__bf16>(stream, f1pad, 32, 32, nullptr, 0, 0, w.f1W1, 32,
                        nullptr, 0, nullptr, nullptr, 0, nullptr, midE, 64,
                        Ee, 64, ACT_NONE);
    launch_gemm<__bf16>(stream, midE, 64, 64, nullptr, 0, 0, w.f1W2, 64,
                        nullptr, 0, nullptr, nullptr, 0, nullptr, fE, 128,
                        Ee, 128, ACT_NONE);
    // conv1: agg = scatter_add(x1[src] * f1); y = Wl*agg + bl + Wr*x1
    pronet_fill<<<fillBlk, 256, 0, stream>>>(aggb, 0.f, aggN);
    pronet_conv_scatter<<<convBlk, 256, 0, stream>>>(
        x1b, fE, esrc, edst, aggb, convTot);
    launch_gemm<float>(stream, aggb, 128, 128, x1b, 128, 128, w.c1Wl, 128,
                       w.c1Wr, 128, w.c1bl, nullptr, 0, yb, nullptr, 128,
                       Nn, 128, ACT_NONE);
    launch_gemm<float>(stream, yb, 128, 128, nullptr, 0, 0, w.l1W, 128,
                       nullptr, 0, w.l1b, nullptr, 0, h1b, nullptr, 128,
                       Nn, 128, ACT_SILU);

    // f2 path (pos_emb, padded K=32), reuses midE/fE/agg/y
    launch_gemm<__bf16>(stream, f2pad, 32, 32, nullptr, 0, 0, w.f2W1, 32,
                        nullptr, 0, nullptr, nullptr, 0, nullptr, midE, 64,
                        Ee, 64, ACT_NONE);
    launch_gemm<__bf16>(stream, midE, 64, 64, nullptr, 0, 0, w.f2W2, 64,
                        nullptr, 0, nullptr, nullptr, 0, nullptr, fE, 128,
                        Ee, 128, ACT_NONE);
    pronet_fill<<<fillBlk, 256, 0, stream>>>(aggb, 0.f, aggN);
    pronet_conv_scatter<<<convBlk, 256, 0, stream>>>(
        x1b, fE, esrc, edst, aggb, convTot);
    launch_gemm<float>(stream, aggb, 128, 128, x1b, 128, 128, w.c2Wl, 128,
                       w.c2Wr, 128, w.c2bl, nullptr, 0, yb, nullptr, 128,
                       Nn, 128, ACT_NONE);
    launch_gemm<float>(stream, yb, 128, 128, nullptr, 0, 0, w.l2W, 128,
                       nullptr, 0, w.l2b, nullptr, 0, h2b, nullptr, 128,
                       Nn, 128, ACT_SILU);

    // lins_cat chain on h = [h1|h2] (dual-A GEMM for the K=256 layer)
    launch_gemm<float>(stream, h1b, 128, 128, h2b, 128, 128,
                       w.cat0W, 256, (const __bf16*)w.cat0W + 128, 256,
                       w.cat0b, nullptr, 0, yb, nullptr, 128, Nn, 128, ACT_SILU);
    launch_gemm<float>(stream, yb, 128, 128, nullptr, 0, 0, w.cat1W, 128,
                       nullptr, 0, w.cat1b, nullptr, 0, aggb, nullptr, 128,
                       Nn, 128, ACT_SILU);
    // third lins_cat layer, then h += x2 (residual applied after activation)
    launch_gemm<float>(stream, aggb, 128, 128, nullptr, 0, 0, w.cat2W, 128,
                       nullptr, 0, w.cat2b, x2b, 128, yb, nullptr, 128,
                       Nn, 128, ACT_SILU);
    // lins: two silu layers
    launch_gemm<float>(stream, yb, 128, 128, nullptr, 0, 0, w.ls0W, 128,
                       nullptr, 0, w.ls0b, nullptr, 0, aggb, nullptr, 128,
                       Nn, 128, ACT_SILU);
    launch_gemm<float>(stream, aggb, 128, 128, nullptr, 0, 0, w.ls1W, 128,
                       nullptr, 0, w.ls1b, nullptr, 0, yb, nullptr, 128,
                       Nn, 128, ACT_SILU);
    // final linear -> next x
    launch_gemm<float>(stream, yb, 128, 128, nullptr, 0, 0, w.finW, 128,
                       nullptr, 0, w.finb, nullptr, 0, xnxt, nullptr, 128,
                       Nn, 128, ACT_NONE);
    float* t = xcur; xcur = xnxt; xnxt = t;
  }

  // ---- epilogue: h = relu(Linear([x|aa])); out = Linear(h) ----
  launch_gemm<float>(stream, xcur, 128, 128, aaBuf, 64, 64,
                     varWb, 192, (const __bf16*)varWb + 128, 192,
                     (const float*)d_in[VARB], nullptr, 0, yb, nullptr, 128,
                     Nn, 128, ACT_RELU);
  pronet_outdot<<<(Nn + 7) / 8, 256, 0, stream>>>(
      yb, (const float*)d_in[OUTW], (const float*)d_in[OUTB],
      (float*)d_out, Nn);
  (void)out_size;
}
